// LSTMModel_81664508166328
// MI455X (gfx1250) — matrix operand order repaired and checked by reading, not compiled, not run on hardware
//
#include <hip/hip_runtime.h>

#define T_SEQ 512
#define BATCH 15
#define BPAD  16
#define HID   1024
#define FOURH 4096
#define VOC   743
#define VPAD  768

typedef __attribute__((ext_vector_type(16))) __bf16 v16bf;
typedef __attribute__((ext_vector_type(8)))  float  v8f;
typedef unsigned short u16;

union V16U { v16bf v; uint4 q[2]; };

__device__ __forceinline__ u16 f2bf(float f) {
  unsigned int u = __float_as_uint(f);
  unsigned int r = u + 0x7FFFu + ((u >> 16) & 1u);  // round-to-nearest-even
  return (u16)(r >> 16);
}

__device__ __forceinline__ v8f splat8(float x) {
  return (v8f){x, x, x, x, x, x, x, x};
}

__device__ __forceinline__ v8f wmma_bf16(v16bf a, v16bf b, v8f c) {
  return __builtin_amdgcn_wmma_f32_16x16x32_bf16(false, a, false, b, (short)0, c, false, false);
}

__device__ __forceinline__ float fast_sig(float x) {   // v_exp_f32 path
  return 1.0f / (1.0f + __expf(-x));
}
__device__ __forceinline__ float fast_tanh(float x) {
  return 1.0f - 2.0f / (__expf(2.0f * x) + 1.0f);
}

// A-matrix (16x32 bf16, MxK) lane layout per CDNA5 ISA 7.12.2:
// lane m = lane%16, half = lane/16; elems 0..7 = K[k0+half*8 .. +7],
// elems 8..15 = K[k0+16+half*8 .. +7]. Two 16B loads, row-major source.
// Works for global (global_load_b128) or LDS-derived (ds_load_b128) pointers.
__device__ __forceinline__ v16bf load_a16(const u16* __restrict__ base, int row, int kh, int k0) {
  V16U u;
  const uint4* p0 = reinterpret_cast<const uint4*>(base + row * HID + k0 + kh * 8);
  const uint4* p1 = reinterpret_cast<const uint4*>(base + row * HID + k0 + 16 + kh * 8);
  u.q[0] = *p0;
  u.q[1] = *p1;
  return u.v;
}

// B-matrix (32x16 bf16, KxN): lane n = lane%16 holds elems 0..7 = K[k0+half*8 .. +7],
// elems 8..15 = K[k0+16+half*8 .. +7] of column n == row n of a [N,K] row-major weight. Two 16B loads.
__device__ __forceinline__ v16bf load_b16(const u16* __restrict__ wrow, int kh, int k0) {
  V16U u;
  const uint4* p = reinterpret_cast<const uint4*>(wrow + k0 + kh * 8);
  u.q[0] = p[0];
  u.q[1] = p[2];
  return u.v;
}

// Async global -> LDS copy, 16B per lane; tracked by ASYNCcnt.
// lds_off is a raw LDS byte offset (dynamic-LDS base == 0, no static LDS here).
__device__ __forceinline__ void async_load_b128(unsigned lds_off, const void* gaddr) {
  asm volatile("global_load_async_to_lds_b128 %0, %1, off"
               :: "v"(lds_off), "v"(gaddr)
               : "memory");
}

__device__ __forceinline__ void wait_async0() {
  asm volatile("s_wait_asynccnt 0x0" ::: "memory");
}

// K=1024 partial GEMM for all 4 gates, FULLY unrolled: straight-line SSA lets
// the scheduler deep-pipeline the 10 loads/k-step against the 4 WMMA chains
// with unique registers (no loop-rotation copies, no branch overhead).
__device__ __forceinline__ void gemm4_full(const u16* __restrict__ s,
                                           const u16* __restrict__ brow,
                                           int n, int kh,
                                           v8f& acc0, v8f& acc1, v8f& acc2, v8f& acc3) {
  const size_t GSTEP = (size_t)HID * HID;  // one gate block of the weight matrix
#pragma unroll
  for (int kk = 0; kk < 32; ++kk) {
    const int k0 = kk << 5;
    if ((kk & 3) == 0) __builtin_prefetch(brow + k0 + 256, 0, 3);
    v16bf a  = load_a16(s, n, kh, k0);
    v16bf b0 = load_b16(brow + 0 * GSTEP, kh, k0);
    v16bf b1 = load_b16(brow + 1 * GSTEP, kh, k0);
    v16bf b2 = load_b16(brow + 2 * GSTEP, kh, k0);
    v16bf b3 = load_b16(brow + 3 * GSTEP, kh, k0);
    acc0 = wmma_bf16(a, b0, acc0);
    acc1 = wmma_bf16(a, b1, acc1);
    acc2 = wmma_bf16(a, b2, acc2);
    acc3 = wmma_bf16(a, b3, acc3);
  }
}

// One LSTM timestep. grid.x = 64 hidden-column tiles, ONE wave per block.
// The wave computes all four gate tiles (i,f,g,o) for its 16 hidden columns:
// A fragment (from LDS) is reused across 4 WMMAs, and the cell update runs
// entirely in registers (identical C/D layout across the four accumulators).
__global__ void __launch_bounds__(32)
lstm_step_kernel(const u16* __restrict__ xin,     // [16,1024] bf16, this step's input
                 const u16* __restrict__ hprev,   // [16,1024] bf16
                 const u16* __restrict__ w_ih,    // [4096,1024] bf16
                 const u16* __restrict__ w_hh,    // [4096,1024] bf16
                 const float* __restrict__ bsum,  // [4096] b_ih+b_hh
                 float* __restrict__ cstate,      // [16,1024] f32 in/out
                 u16* __restrict__ yout,          // [16,1024] bf16 h out (next-step input)
                 float* __restrict__ hf32)        // [16,1024] f32 h (for hn output)
{
  extern __shared__ char smem[];  // [0,32KB) x_t   [32KB,64KB) h_prev
  const int lane = threadIdx.x & 31;
  const int n    = lane & 15;
  const int kh   = lane >> 4;
  const int j0   = blockIdx.x << 4;

  // ---- stage x_t and h_prev into LDS with async-to-LDS (64KB total) ----
  {
    const char* gx = (const char*)xin;
    const char* gh = (const char*)hprev;
    for (int it = 0; it < 64; ++it) {
      const unsigned o = (unsigned)(it * 512 + lane * 16);
      async_load_b128(o, gx + o);
      async_load_b128(32768u + o, gh + o);
    }
  }

  const u16* brow_ih = w_ih + (size_t)(j0 + n) * HID;  // gate g row at + g*HID*HID
  const u16* brow_hh = w_hh + (size_t)(j0 + n) * HID;

  v8f acc0 = splat8(bsum[0 * HID + j0 + n]);  // i
  v8f acc1 = splat8(bsum[1 * HID + j0 + n]);  // f
  v8f acc2 = splat8(bsum[2 * HID + j0 + n]);  // g
  v8f acc3 = splat8(bsum[3 * HID + j0 + n]);  // o

  wait_async0();  // LDS staging complete (single wave -> no barrier needed)

  const u16* sx = (const u16*)smem;
  const u16* sh = (const u16*)(smem + 32768);

  gemm4_full(sx, brow_ih, n, kh, acc0, acc1, acc2, acc3);  // x_t @ W_ih^T
  gemm4_full(sh, brow_hh, n, kh, acc0, acc1, acc2, acc3);  // h_prev @ W_hh^T

  // ---- fused cell update, all in registers ----
  // C/D layout: lane half kh holds rows kh*8+v, column n.
#pragma unroll
  for (int v = 0; v < 8; ++v) {
    const int mm  = kh * 8 + v;
    const int idx = mm * HID + j0 + n;
    const float iv = fast_sig(acc0[v]);
    const float fv = fast_sig(acc1[v]);
    const float ov = fast_sig(acc3[v]);
    const float cn = fv * cstate[idx] + iv * fast_tanh(acc2[v]);
    const float hn = ov * fast_tanh(cn);
    cstate[idx] = cn;
    hf32[idx]   = hn;
    yout[idx]   = f2bf(hn);
  }
}

// logits[b,t,v] = y1[t,b,:] . fc_w[v,:] + fc_b[v]
__global__ void __launch_bounds__(256)
fc_kernel(const u16* __restrict__ y1,   // [T,16,1024] bf16
          const u16* __restrict__ fcw,  // [768,1024] bf16 zero-padded
          const float* __restrict__ fcb,
          float* __restrict__ out)      // logits region of d_out
{
  const int lane = threadIdx.x & 31;
  const int wv   = threadIdx.x >> 5;
  const int t    = blockIdx.x;
  const int ct   = blockIdx.y * 8 + wv;   // 0..47 column tiles
  const int n    = lane & 15;
  const int kh   = lane >> 4;
  const int vcol = ct * 16 + n;
  const u16* A    = y1 + (size_t)t * (BPAD * HID);
  const u16* brow = fcw + (size_t)vcol * HID;
  const float bias = (vcol < VOC) ? fcb[vcol] : 0.0f;
  v8f acc = splat8(bias);
#pragma unroll
  for (int kk = 0; kk < 32; ++kk) {
    const int k0 = kk << 5;
    if ((kk & 3) == 0) __builtin_prefetch(brow + k0 + 256, 0, 3);
    v16bf a = load_a16(A, n, kh, k0);
    v16bf b = load_b16(brow, kh, k0);
    acc = wmma_bf16(a, b, acc);
  }
  if (vcol < VOC) {
#pragma unroll
    for (int v = 0; v < 8; ++v) {
      const int mm = kh * 8 + v;  // batch row
      if (mm < BATCH)
        out[(size_t)mm * (T_SEQ * VOC) + (size_t)t * VOC + vcol] = acc[v];
    }
  }
}

__global__ void f32_to_bf16_kernel(const float* __restrict__ src, u16* __restrict__ dst, int n) {
  for (int i = blockIdx.x * blockDim.x + threadIdx.x; i < n; i += gridDim.x * blockDim.x)
    dst[i] = f2bf(src[i]);
}

__global__ void fcw_convert_kernel(const float* __restrict__ src, u16* __restrict__ dst) {
  const int n = VPAD * HID;
  for (int i = blockIdx.x * blockDim.x + threadIdx.x; i < n; i += gridDim.x * blockDim.x) {
    const int r = i >> 10;
    dst[i] = (r < VOC) ? f2bf(src[i]) : (u16)0;
  }
}

__global__ void bias_sum_kernel(const float* __restrict__ a, const float* __restrict__ b,
                                float* __restrict__ o) {
  const int i = blockIdx.x * blockDim.x + threadIdx.x;
  if (i < FOURH) o[i] = a[i] + b[i];
}

__global__ void init_state_kernel(const float* __restrict__ h0, const float* __restrict__ c0,
                                  u16* __restrict__ hb, float* __restrict__ hf,
                                  float* __restrict__ cf) {
  for (int i = blockIdx.x * blockDim.x + threadIdx.x; i < BPAD * HID;
       i += gridDim.x * blockDim.x) {
    const int m = i >> 10;
    const float hv = (m < BATCH) ? h0[i] : 0.0f;
    const float cv = (m < BATCH) ? c0[i] : 0.0f;
    hb[i] = f2bf(hv);
    hf[i] = hv;
    cf[i] = cv;
  }
}

__global__ void embed_kernel(const int* __restrict__ x, const float* __restrict__ emb,
                             u16* __restrict__ xe) {
  const int t = blockIdx.x;
  const int m = blockIdx.y;
  u16* dst = xe + ((size_t)t * BPAD + m) * HID;
  if (m >= BATCH) {
    for (int e = threadIdx.x; e < HID; e += blockDim.x) dst[e] = 0;
    return;
  }
  const int idx = x[m * T_SEQ + t];
  const float* src = emb + (size_t)idx * HID;
  for (int e = threadIdx.x; e < HID; e += blockDim.x) dst[e] = f2bf(src[e]);
}

__global__ void finalize_kernel(const float* __restrict__ hf0, const float* __restrict__ hf1,
                                const float* __restrict__ c0f, const float* __restrict__ c1f,
                                float* __restrict__ out) {
  const int HN_OFF = BATCH * T_SEQ * VOC;
  const int CN_OFF = HN_OFF + 2 * BATCH * HID;
  for (int i = blockIdx.x * blockDim.x + threadIdx.x; i < BATCH * HID;
       i += gridDim.x * blockDim.x) {
    out[HN_OFF + i]               = hf0[i];
    out[HN_OFF + BATCH * HID + i] = hf1[i];
    out[CN_OFF + i]               = c0f[i];
    out[CN_OFF + BATCH * HID + i] = c1f[i];
  }
}

extern "C" void kernel_launch(void* const* d_in, const int* in_sizes, int n_in,
                              void* d_out, int out_size, void* d_ws, size_t ws_size,
                              hipStream_t stream) {
  (void)in_sizes; (void)n_in; (void)out_size; (void)ws_size;
  const int*   x     = (const int*)  d_in[0];
  const float* h0    = (const float*)d_in[1];
  const float* c0    = (const float*)d_in[2];
  const float* emb   = (const float*)d_in[3];
  const float* w_ih0 = (const float*)d_in[4];
  const float* w_hh0 = (const float*)d_in[5];
  const float* b_ih0 = (const float*)d_in[6];
  const float* b_hh0 = (const float*)d_in[7];
  const float* w_ih1 = (const float*)d_in[8];
  const float* w_hh1 = (const float*)d_in[9];
  const float* b_ih1 = (const float*)d_in[10];
  const float* b_hh1 = (const float*)d_in[11];
  const float* fc_w  = (const float*)d_in[12];
  const float* fc_b  = (const float*)d_in[13];
  float* out = (float*)d_out;

  char* ws = (char*)d_ws;
  size_t off = 0;
  auto alloc = [&](size_t bytes) -> char* {
    char* p = ws + off;
    off += (bytes + 255) & ~(size_t)255;
    return p;
  };
  const size_t WELEMS = (size_t)FOURH * HID;
  const size_t SLICE  = (size_t)BPAD * HID;  // one timestep slice, elems
  u16*   Wih0b = (u16*)alloc(WELEMS * 2);
  u16*   Whh0b = (u16*)alloc(WELEMS * 2);
  u16*   Wih1b = (u16*)alloc(WELEMS * 2);
  u16*   Whh1b = (u16*)alloc(WELEMS * 2);
  u16*   Fcwb  = (u16*)alloc((size_t)VPAD * HID * 2);
  float* Bs0   = (float*)alloc(FOURH * 4);
  float* Bs1   = (float*)alloc(FOURH * 4);
  u16*   XE    = (u16*)alloc((size_t)T_SEQ * SLICE * 2);
  u16*   Y0    = (u16*)alloc((size_t)T_SEQ * SLICE * 2);
  u16*   Y1    = (u16*)alloc((size_t)T_SEQ * SLICE * 2);
  u16*   Hi0   = (u16*)alloc(SLICE * 2);
  u16*   Hi1   = (u16*)alloc(SLICE * 2);
  float* Hf0   = (float*)alloc(SLICE * 4);
  float* Hf1   = (float*)alloc(SLICE * 4);
  float* C0s   = (float*)alloc(SLICE * 4);
  float* C1s   = (float*)alloc(SLICE * 4);

  // ---- prep: bf16 weights, fused biases, init state, embedding gather ----
  f32_to_bf16_kernel<<<2048, 256, 0, stream>>>(w_ih0, Wih0b, (int)WELEMS);
  f32_to_bf16_kernel<<<2048, 256, 0, stream>>>(w_hh0, Whh0b, (int)WELEMS);
  f32_to_bf16_kernel<<<2048, 256, 0, stream>>>(w_ih1, Wih1b, (int)WELEMS);
  f32_to_bf16_kernel<<<2048, 256, 0, stream>>>(w_hh1, Whh1b, (int)WELEMS);
  fcw_convert_kernel<<<512, 256, 0, stream>>>(fc_w, Fcwb);
  bias_sum_kernel<<<16, 256, 0, stream>>>(b_ih0, b_hh0, Bs0);
  bias_sum_kernel<<<16, 256, 0, stream>>>(b_ih1, b_hh1, Bs1);
  init_state_kernel<<<64, 256, 0, stream>>>(h0, c0, Hi0, Hf0, C0s);
  init_state_kernel<<<64, 256, 0, stream>>>(h0 + BATCH * HID, c0 + BATCH * HID, Hi1, Hf1, C1s);
  embed_kernel<<<dim3(T_SEQ, BPAD), 256, 0, stream>>>(x, emb, XE);

  const size_t SHBYTES = 65536;  // x_t (32KB) + h_prev (32KB) LDS staging

  // ---- layer 0 recurrence: stream order carries the t -> t+1 dependency ----
  for (int t = 0; t < T_SEQ; ++t) {
    const u16* hp = (t == 0) ? Hi0 : (Y0 + (size_t)(t - 1) * SLICE);
    lstm_step_kernel<<<64, 32, SHBYTES, stream>>>(XE + (size_t)t * SLICE, hp, Wih0b, Whh0b,
                                                  Bs0, C0s, Y0 + (size_t)t * SLICE, Hf0);
  }
  // ---- layer 1 recurrence ----
  for (int t = 0; t < T_SEQ; ++t) {
    const u16* hp = (t == 0) ? Hi1 : (Y1 + (size_t)(t - 1) * SLICE);
    lstm_step_kernel<<<64, 32, SHBYTES, stream>>>(Y0 + (size_t)t * SLICE, hp, Wih1b, Whh1b,
                                                  Bs1, C1s, Y1 + (size_t)t * SLICE, Hf1);
  }
  // ---- output projection + final states ----
  fc_kernel<<<dim3(T_SEQ, 6), 256, 0, stream>>>(Y1, Fcwb, fc_b, out);
  finalize_kernel<<<60, 256, 0, stream>>>(Hf0, Hf1, C0s, C1s, out);
}
